// LinearColumnwise_18708877542041
// MI455X (gfx1250) — matrix operand order repaired and checked by reading, not compiled, not run on hardware
//
#include <hip/hip_runtime.h>
#include <hip/hip_bf16.h>

// C[16384,4096] = A[16384,4096] @ W[4096,4096] + bias   (fp32 io)
//
// Two-pass split-bf16 WMMA GEMM for gfx1250:
//   Pass 1: stream A and W once, split each fp32 into (hi, lo) bf16 and store
//           into d_ws in K-blocked tile layout ([kb][row][32], W transposed),
//           so pass-2 tiles are contiguous 8 KB blocks.
//   Pass 2: double-buffered GEMM; tiles staged with GLOBAL_LOAD_ASYNC_TO_LDS
//           (ASYNCcnt) overlapped with v_wmma_f32_16x16x32_bf16 math.
//           Product = hi*hi + hi*lo + lo*hi, fp32 accumulate (~fp32 accuracy).
// Falls back to the fused single-pass kernel if ws_size is too small.

typedef __attribute__((ext_vector_type(16))) __bf16 v16bf;
typedef __attribute__((ext_vector_type(8)))  __bf16 v8bf;
typedef __attribute__((ext_vector_type(8)))  float  v8f;

#define M_TOT 16384
#define K_DIM 4096
#define N_DIM 4096

#define BM 128
#define BN 128
#define BK 32
#define KPAD 40   // padded LDS row length in 16-bit elems (80B, 16B-aligned)
#define NKB (K_DIM / BK)   // 128 K-blocks

// fp32 -> (hi, lo) bf16 split with round-to-nearest-even, via integer ops.
__device__ __forceinline__ void split_bf16(float x, unsigned short& h, unsigned short& l) {
  unsigned int u = __float_as_uint(x);
  unsigned int hr = u + 0x7FFFu + ((u >> 16) & 1u);
  h = (unsigned short)(hr >> 16);
  float hf = __uint_as_float(((unsigned int)h) << 16);
  float r = x - hf;
  unsigned int ur = __float_as_uint(r);
  unsigned int lr = ur + 0x7FFFu + ((ur >> 16) & 1u);
  l = (unsigned short)(lr >> 16);
}

__device__ __forceinline__ v16bf join16(v8bf a, v8bf b) {
  v16bf r;
#pragma unroll
  for (int e = 0; e < 8; ++e) { r[e] = a[e]; r[e + 8] = b[e]; }
  return r;
}

// ---------------------------------------------------------------------------
// Pass 1a: split A[16384,4096] fp32 -> Ahi/Alo bf16, K-blocked [kb][m][32].
// ---------------------------------------------------------------------------
__global__ __launch_bounds__(256)
void split_A_kernel(const float* __restrict__ A,
                    unsigned short* __restrict__ Ahi,
                    unsigned short* __restrict__ Alo)
{
  const size_t f = (size_t)blockIdx.x * 256 + threadIdx.x;  // float4 slot
  const size_t m = f >> 10;                 // 1024 float4 per row
  const int    k = (int)(f & 1023) << 2;
  const int    kb = k >> 5, kin = k & 31;

  const float4 v = *reinterpret_cast<const float4*>(A + m * K_DIM + k);
  const float vv[4] = {v.x, v.y, v.z, v.w};
  unsigned short h[4], l[4];
#pragma unroll
  for (int j = 0; j < 4; ++j) split_bf16(vv[j], h[j], l[j]);

  const size_t o = ((size_t)kb * M_TOT + m) * BK + kin;
  *reinterpret_cast<ushort4*>(Ahi + o) = make_ushort4(h[0], h[1], h[2], h[3]);
  *reinterpret_cast<ushort4*>(Alo + o) = make_ushort4(l[0], l[1], l[2], l[3]);
}

// ---------------------------------------------------------------------------
// Pass 1b: split + transpose W[4096,4096] fp32 -> Whi/Wlo bf16 [kb][n][32].
// One block per (kb, n-block of 128); transpose through LDS.
// ---------------------------------------------------------------------------
__global__ __launch_bounds__(256)
void split_W_kernel(const float* __restrict__ W,
                    unsigned short* __restrict__ Whi,
                    unsigned short* __restrict__ Wlo)
{
  __shared__ unsigned short hi[BN * KPAD];
  __shared__ unsigned short lo[BN * KPAD];

  const int tid = threadIdx.x;
  const int nb  = blockIdx.x & 31;   // N/BN = 32
  const int kb  = blockIdx.x >> 5;   // K/BK = 128
  const int n0  = nb * BN;
  const int k0  = kb * BK;

#pragma unroll
  for (int i = 0; i < 4; ++i) {
    const int f    = tid + i * 256;    // 1024 float4 slots (32 rows x 32/row)
    const int kr   = f >> 5;
    const int ncol = (f & 31) << 2;
    const float4 v = *reinterpret_cast<const float4*>(
        W + (size_t)(k0 + kr) * N_DIM + (n0 + ncol));
    const float vv[4] = {v.x, v.y, v.z, v.w};
#pragma unroll
    for (int j = 0; j < 4; ++j) {
      unsigned short h, l;
      split_bf16(vv[j], h, l);
      hi[(ncol + j) * KPAD + kr] = h;
      lo[(ncol + j) * KPAD + kr] = l;
    }
  }
  __syncthreads();

  // coalesced 16B writes of the transposed tile
#pragma unroll
  for (int i = 0; i < 2; ++i) {
    const int c   = tid + (i << 8);    // 512 chunks of 16B
    const int row = c >> 2;
    const int ch  = c & 3;
    const size_t o = ((size_t)kb * N_DIM + n0 + row) * BK + ch * 8;
    *reinterpret_cast<uint4*>(Whi + o) =
        *reinterpret_cast<const uint4*>(&hi[row * KPAD + ch * 8]);
    *reinterpret_cast<uint4*>(Wlo + o) =
        *reinterpret_cast<const uint4*>(&lo[row * KPAD + ch * 8]);
  }
}

// ---------------------------------------------------------------------------
// Pass 2: double-buffered WMMA GEMM, async global->LDS tile staging.
// ---------------------------------------------------------------------------
__global__ __launch_bounds__(256)
void gemm_ws_wmma(const unsigned short* __restrict__ Ahi,
                  const unsigned short* __restrict__ Alo,
                  const unsigned short* __restrict__ Whi,
                  const unsigned short* __restrict__ Wlo,
                  const float* __restrict__ bias,
                  float* __restrict__ out)
{
  // [stage][array][tile]; arrays: 0=Ahi 1=Alo 2=Bhi 3=Blo; rows padded to 80B.
  __shared__ unsigned short smem[2 * 4 * BM * KPAD];   // 80 KB

  const int tid  = threadIdx.x;
  const int lane = tid & 31;
  const int wave = tid >> 5;
  const int wr   = wave >> 2;      // 0..1
  const int wc   = wave & 3;       // 0..3

  const int nb = blockIdx.x & 31;
  const int mb = blockIdx.x >> 5;
  const int m0 = mb * BM;
  const int n0 = nb * BN;

  // per-thread staging chunks: 512 16B chunks per array, 2 per thread
  const int c0  = tid;
  const int c1  = tid + 256;
  const unsigned go0 = (unsigned)(((c0 >> 2) << 6) + ((c0 & 3) << 4)); // row*64+ch*16
  const unsigned go1 = (unsigned)(((c1 >> 2) << 6) + ((c1 & 3) << 4));
  const unsigned lo0 = (unsigned)((c0 >> 2) * (KPAD * 2) + ((c0 & 3) << 4));
  const unsigned lo1 = (unsigned)((c1 >> 2) * (KPAD * 2) + ((c1 & 3) << 4));

  const unsigned long long gA[2] = {(unsigned long long)(uintptr_t)Ahi,
                                    (unsigned long long)(uintptr_t)Alo};
  const unsigned long long gB[2] = {(unsigned long long)(uintptr_t)Whi,
                                    (unsigned long long)(uintptr_t)Wlo};

  auto stage = [&](int s, int kb) {
    const unsigned long long aoff = ((unsigned long long)kb * M_TOT + m0) * (BK * 2);
    const unsigned long long boff = ((unsigned long long)kb * N_DIM + n0) * (BK * 2);
#pragma unroll
    for (int arr = 0; arr < 4; ++arr) {
      const unsigned long long gbase =
          (arr < 2) ? (gA[arr] + aoff) : (gB[arr - 2] + boff);
      const unsigned lbase =
          (unsigned)(uintptr_t)(smem + ((s * 4 + arr) * BM * KPAD));
      asm volatile("global_load_async_to_lds_b128 %0, %1, %2"
                   :: "v"(lbase + lo0), "v"(go0), "s"(gbase) : "memory");
      asm volatile("global_load_async_to_lds_b128 %0, %1, %2"
                   :: "v"(lbase + lo1), "v"(go1), "s"(gbase) : "memory");
    }
  };

  v8f acc[4][2];
#pragma unroll
  for (int i = 0; i < 4; ++i)
#pragma unroll
    for (int j = 0; j < 2; ++j)
      acc[i][j] = v8f{};

  // CDNA5 WMMA fragment addressing (see ISA 7.12.2):
  const int arow  = wr * 64 + (lane & 15);
  const int akoff = (lane & 16) ? 16 : 0;   // bytes
  const int bcol  = wc * 32 + (lane & 15);
  const int bkoff = (lane & 16) ? 16 : 0;   // bytes

  stage(0, 0);

  int s = 0;
  for (int kb = 0; kb < NKB; ++kb) {
    asm volatile("s_wait_asynccnt 0" ::: "memory");
    __syncthreads();

    if (kb + 1 < NKB) stage(s ^ 1, kb + 1);

    const unsigned short* sAh = smem + ((s * 4 + 0) * BM * KPAD);
    const unsigned short* sAl = smem + ((s * 4 + 1) * BM * KPAD);
    const unsigned short* sBh = smem + ((s * 4 + 2) * BM * KPAD);
    const unsigned short* sBl = smem + ((s * 4 + 3) * BM * KPAD);

    v16bf ah[4], al[4];
#pragma unroll
    for (int fm = 0; fm < 4; ++fm) {
      const char* ph = (const char*)(sAh + (arow + fm * 16) * KPAD) + akoff;
      const char* pl = (const char*)(sAl + (arow + fm * 16) * KPAD) + akoff;
      ah[fm] = join16(*(const v8bf*)ph, *(const v8bf*)(ph + 32));
      al[fm] = join16(*(const v8bf*)pl, *(const v8bf*)(pl + 32));
    }

#pragma unroll
    for (int fn = 0; fn < 2; ++fn) {
      const char* ph = (const char*)(sBh + (bcol + fn * 16) * KPAD) + bkoff;
      const char* pl = (const char*)(sBl + (bcol + fn * 16) * KPAD) + bkoff;
      const v16bf bh = join16(*(const v8bf*)ph, *(const v8bf*)(ph + 32));
      const v16bf bl = join16(*(const v8bf*)pl, *(const v8bf*)(pl + 32));
#pragma unroll
      for (int fm = 0; fm < 4; ++fm) {
        acc[fm][fn] = __builtin_amdgcn_wmma_f32_16x16x32_bf16(
            false, ah[fm], false, bh, (short)0, acc[fm][fn], false, false);
        acc[fm][fn] = __builtin_amdgcn_wmma_f32_16x16x32_bf16(
            false, ah[fm], false, bl, (short)0, acc[fm][fn], false, false);
        acc[fm][fn] = __builtin_amdgcn_wmma_f32_16x16x32_bf16(
            false, al[fm], false, bh, (short)0, acc[fm][fn], false, false);
      }
    }

    __syncthreads();
    s ^= 1;
  }

  // epilogue: C/D layout -> VGPR i: M=i (lanes 0-15) / M=i+8 (lanes 16-31)
  const int half = lane >> 4;
#pragma unroll
  for (int fn = 0; fn < 2; ++fn) {
    const int n = n0 + wc * 32 + fn * 16 + (lane & 15);
    const float b = bias[n];
#pragma unroll
    for (int fm = 0; fm < 4; ++fm) {
      const int mbase = m0 + wr * 64 + fm * 16 + half * 8;
#pragma unroll
      for (int r = 0; r < 8; ++r) {
        out[(size_t)(mbase + r) * N_DIM + n] = acc[fm][fn][r] + b;
      }
    }
  }
}

// ---------------------------------------------------------------------------
// Fallback: fused single-pass kernel (round-1), used when ws is too small.
// ---------------------------------------------------------------------------
__global__ __launch_bounds__(256)
void gemm_bias_bf16x3_wmma(const float* __restrict__ A,
                           const float* __restrict__ W,
                           const float* __restrict__ bias,
                           float* __restrict__ out)
{
  __shared__ unsigned short Ahi[BM * KPAD];
  __shared__ unsigned short Alo[BM * KPAD];
  __shared__ unsigned short Bhi[BN * KPAD];
  __shared__ unsigned short Blo[BN * KPAD];

  const int tid  = threadIdx.x;
  const int lane = tid & 31;
  const int wave = tid >> 5;
  const int wr   = wave >> 2;
  const int wc   = wave & 3;

  const int nb = blockIdx.x & 31;
  const int mb = blockIdx.x >> 5;
  const int m0 = mb * BM;
  const int n0 = nb * BN;

  v8f acc[4][2];
#pragma unroll
  for (int i = 0; i < 4; ++i)
#pragma unroll
    for (int j = 0; j < 2; ++j)
      acc[i][j] = v8f{};

  const int arow  = wr * 64 + (lane & 15);
  const int akoff = (lane & 16) ? 16 : 0;
  const int bcol  = wc * 32 + (lane & 15);
  const int bkoff = (lane & 16) ? 16 : 0;

  for (int kt = 0; kt < K_DIM; kt += BK) {
#pragma unroll
    for (int i = 0; i < 4; ++i) {
      const int f = tid + i * 256;
      {
        const int row = f >> 3;
        const int kc  = (f & 7) << 2;
        const float4 v = *reinterpret_cast<const float4*>(
            A + (size_t)(m0 + row) * K_DIM + (kt + kc));
        const float vv[4] = {v.x, v.y, v.z, v.w};
        unsigned short h[4], l[4];
#pragma unroll
        for (int j = 0; j < 4; ++j) split_bf16(vv[j], h[j], l[j]);
        *reinterpret_cast<ushort4*>(&Ahi[row * KPAD + kc]) =
            make_ushort4(h[0], h[1], h[2], h[3]);
        *reinterpret_cast<ushort4*>(&Alo[row * KPAD + kc]) =
            make_ushort4(l[0], l[1], l[2], l[3]);
      }
      {
        const int kr   = f >> 5;
        const int ncol = (f & 31) << 2;
        const float4 v = *reinterpret_cast<const float4*>(
            W + (size_t)(kt + kr) * N_DIM + (n0 + ncol));
        const float vv[4] = {v.x, v.y, v.z, v.w};
#pragma unroll
        for (int j = 0; j < 4; ++j) {
          unsigned short h, l;
          split_bf16(vv[j], h, l);
          Bhi[(ncol + j) * KPAD + kr] = h;
          Blo[(ncol + j) * KPAD + kr] = l;
        }
      }
    }
    if (kt + BK < K_DIM) {
      __builtin_prefetch(A + (size_t)(m0 + (tid >> 3)) * K_DIM +
                             (kt + BK + ((tid & 7) << 2)), 0, 0);
      __builtin_prefetch(W + (size_t)(kt + BK + (tid >> 5)) * N_DIM +
                             (n0 + ((tid & 31) << 2)), 0, 0);
    }
    __syncthreads();

    v16bf ah[4], al[4];
#pragma unroll
    for (int fm = 0; fm < 4; ++fm) {
      const char* ph = (const char*)(Ahi + (arow + fm * 16) * KPAD) + akoff;
      const char* pl = (const char*)(Alo + (arow + fm * 16) * KPAD) + akoff;
      ah[fm] = join16(*(const v8bf*)ph, *(const v8bf*)(ph + 32));
      al[fm] = join16(*(const v8bf*)pl, *(const v8bf*)(pl + 32));
    }
#pragma unroll
    for (int fn = 0; fn < 2; ++fn) {
      const char* ph = (const char*)(Bhi + (bcol + fn * 16) * KPAD) + bkoff;
      const char* pl = (const char*)(Blo + (bcol + fn * 16) * KPAD) + bkoff;
      const v16bf bh = join16(*(const v8bf*)ph, *(const v8bf*)(ph + 32));
      const v16bf bl = join16(*(const v8bf*)pl, *(const v8bf*)(pl + 32));
#pragma unroll
      for (int fm = 0; fm < 4; ++fm) {
        acc[fm][fn] = __builtin_amdgcn_wmma_f32_16x16x32_bf16(
            false, ah[fm], false, bh, (short)0, acc[fm][fn], false, false);
        acc[fm][fn] = __builtin_amdgcn_wmma_f32_16x16x32_bf16(
            false, ah[fm], false, bl, (short)0, acc[fm][fn], false, false);
        acc[fm][fn] = __builtin_amdgcn_wmma_f32_16x16x32_bf16(
            false, al[fm], false, bh, (short)0, acc[fm][fn], false, false);
      }
    }
    __syncthreads();
  }

  const int half = lane >> 4;
#pragma unroll
  for (int fn = 0; fn < 2; ++fn) {
    const int n = n0 + wc * 32 + fn * 16 + (lane & 15);
    const float b = bias[n];
#pragma unroll
    for (int fm = 0; fm < 4; ++fm) {
      const int mbase = m0 + wr * 64 + fm * 16 + half * 8;
#pragma unroll
      for (int r = 0; r < 8; ++r) {
        out[(size_t)(mbase + r) * N_DIM + n] = acc[fm][fn][r] + b;
      }
    }
  }
}

extern "C" void kernel_launch(void* const* d_in, const int* in_sizes, int n_in,
                              void* d_out, int out_size, void* d_ws, size_t ws_size,
                              hipStream_t stream) {
  const float* A    = (const float*)d_in[0];   // [8,2048,4096] == [16384,4096]
  const float* W    = (const float*)d_in[1];   // [4096,4096]
  const float* bias = (const float*)d_in[2];   // [4096]
  float* out        = (float*)d_out;           // [16384,4096]

  const size_t SA = (size_t)M_TOT * K_DIM * sizeof(unsigned short); // 128 MiB
  const size_t SW = (size_t)K_DIM * N_DIM * sizeof(unsigned short); //  32 MiB
  const size_t NEED = 2 * SA + 2 * SW;                              // 320 MiB

  const int grid = (M_TOT / BM) * (N_DIM / BN);  // 4096 blocks

  if (ws_size >= NEED) {
    unsigned short* Ahi = (unsigned short*)d_ws;
    unsigned short* Alo = (unsigned short*)((char*)d_ws + SA);
    unsigned short* Whi = (unsigned short*)((char*)d_ws + 2 * SA);
    unsigned short* Wlo = (unsigned short*)((char*)d_ws + 2 * SA + SW);

    const int gA = (int)(((size_t)M_TOT * K_DIM / 4) / 256);  // 65536 blocks
    split_A_kernel<<<gA, 256, 0, stream>>>(A, Ahi, Alo);
    split_W_kernel<<<(K_DIM / BK) * (N_DIM / BN), 256, 0, stream>>>(W, Whi, Wlo);
    gemm_ws_wmma<<<grid, 256, 0, stream>>>(Ahi, Alo, Whi, Wlo, bias, out);
  } else {
    gemm_bias_bf16x3_wmma<<<grid, 256, 0, stream>>>(A, W, bias, out);
  }
}
